// TernaryLinear_13151189860577
// MI455X (gfx1250) — 8-bit matrix operand order repaired and checked by reading, compiled, not run on hardware
//
#include <hip/hip_runtime.h>

#define IN_F 4096
#define OUT_F 4096
#define QMAX 127.0f
#define TEPS 1e-10f

typedef __attribute__((ext_vector_type(8))) int v8i;

__global__ __launch_bounds__(256) void quantize_act(const float* __restrict__ x,
                                                    signed char* __restrict__ xq,
                                                    float* __restrict__ act_scale) {
    __shared__ float red[256];
    const int row = blockIdx.x;
    const float* xr = x + (size_t)row * IN_F;

    float4 v[4];
    float m = 0.0f;
#pragma unroll
    for (int i = 0; i < 4; ++i) {
        v[i] = ((const float4*)xr)[threadIdx.x + 256 * i];
        m = fmaxf(m, fmaxf(fmaxf(fabsf(v[i].x), fabsf(v[i].y)),
                           fmaxf(fabsf(v[i].z), fabsf(v[i].w))));
    }
    red[threadIdx.x] = m;
    __syncthreads();
#pragma unroll
    for (int s = 128; s > 0; s >>= 1) {
        if (threadIdx.x < s) red[threadIdx.x] = fmaxf(red[threadIdx.x], red[threadIdx.x + s]);
        __syncthreads();
    }
    const float scale = fmaxf(red[0], TEPS) / QMAX;
    const float inv = QMAX / fmaxf(red[0], TEPS);
    if (threadIdx.x == 0) act_scale[row] = scale;

    int* xqr = (int*)(xq + (size_t)row * IN_F);
#pragma unroll
    for (int i = 0; i < 4; ++i) {
        int q0 = (int)rintf(v[i].x * inv);
        int q1 = (int)rintf(v[i].y * inv);
        int q2 = (int)rintf(v[i].z * inv);
        int q3 = (int)rintf(v[i].w * inv);
        q0 = q0 > 127 ? 127 : (q0 < -127 ? -127 : q0);
        q1 = q1 > 127 ? 127 : (q1 < -127 ? -127 : q1);
        q2 = q2 > 127 ? 127 : (q2 < -127 ? -127 : q2);
        q3 = q3 > 127 ? 127 : (q3 < -127 ? -127 : q3);
        xqr[threadIdx.x + 256 * i] =
            (q0 & 255) | ((q1 & 255) << 8) | ((q2 & 255) << 16) | ((q3 & 255) << 24);
    }
}

__global__ __launch_bounds__(256) void pack_w(const float* __restrict__ w,
                                              signed char* __restrict__ w8, int n4) {
    const int idx = blockIdx.x * 256 + threadIdx.x;
    if (idx >= n4) return;
    const float4 v = ((const float4*)w)[idx];
    const int q0 = (int)rintf(v.x);
    const int q1 = (int)rintf(v.y);
    const int q2 = (int)rintf(v.z);
    const int q3 = (int)rintf(v.w);
    ((int*)w8)[idx] = (q0 & 255) | ((q1 & 255) << 8) | ((q2 & 255) << 16) | ((q3 & 255) << 24);
}

__global__ __launch_bounds__(256) void ternary_gemm(const signed char* __restrict__ xq,
                                                    const signed char* __restrict__ w8,
                                                    const float* __restrict__ act_scale,
                                                    const float* __restrict__ w_scale_p,
                                                    const float* __restrict__ bias,
                                                    float* __restrict__ out, int nblk) {
    const int lane = threadIdx.x & 31;
    const int wave = threadIdx.x >> 5;
    const int wid = blockIdx.x * 8 + wave;
    const int bn = wid % nblk;
    const int bm = wid / nblk;

    const int half = lane >> 4;
    const int l16 = lane & 15;

    const signed char* ap0 = xq + (size_t)(bm * 32 + l16) * IN_F + half * 16;
    const signed char* bp0 = w8 + (size_t)(bn * 64 + l16) * IN_F + half * 16;

    v8i acc[2][4];
#pragma unroll
    for (int i = 0; i < 2; ++i)
#pragma unroll
        for (int j = 0; j < 4; ++j) acc[i][j] = (v8i){0, 0, 0, 0, 0, 0, 0, 0};

#pragma unroll 2
    for (int k = 0; k < IN_F; k += 64) {
        __builtin_prefetch(ap0 + k + 256, 0, 3);
        __builtin_prefetch(ap0 + (size_t)16 * IN_F + k + 256, 0, 3);
#pragma unroll
        for (int j = 0; j < 4; ++j)
            __builtin_prefetch(bp0 + (size_t)(j * 16) * IN_F + k + 256, 0, 3);

        v8i A[2], B[4];
#pragma unroll
        for (int i = 0; i < 2; ++i) {
            const signed char* p = ap0 + (size_t)(i * 16) * IN_F + k;
            const int4 t0 = *(const int4*)(p);
            const int4 t1 = *(const int4*)(p + 32);
            A[i] = (v8i){t0.x, t0.y, t0.z, t0.w, t1.x, t1.y, t1.z, t1.w};
        }
#pragma unroll
        for (int j = 0; j < 4; ++j) {
            const signed char* p = bp0 + (size_t)(j * 16) * IN_F + k;
            const int4 u0 = *(const int4*)(p);
            const int4 u1 = *(const int4*)(p + 32);
            B[j] = (v8i){u0.x, u0.y, u0.z, u0.w, u1.x, u1.y, u1.z, u1.w};
        }
#pragma unroll
        for (int i = 0; i < 2; ++i)
#pragma unroll
            for (int j = 0; j < 4; ++j)
                acc[i][j] = __builtin_amdgcn_wmma_i32_16x16x64_iu8(true, A[i], true, B[j],
                                                                   acc[i][j], false, false);
    }

    const float wsv = *w_scale_p;
    float bc[4];
    int ncol[4];
#pragma unroll
    for (int j = 0; j < 4; ++j) {
        ncol[j] = bn * 64 + j * 16 + l16;
        bc[j] = bias[ncol[j]];
    }
#pragma unroll
    for (int i = 0; i < 2; ++i) {
        const int mbase = bm * 32 + i * 16 + half * 8;
        float* orow = out + (size_t)mbase * OUT_F;
#pragma unroll
        for (int r = 0; r < 8; ++r) {
            const float s = act_scale[mbase + r] * wsv;
#pragma unroll
            for (int j = 0; j < 4; ++j)
                orow[(size_t)r * OUT_F + ncol[j]] = (float)acc[i][j][r] * s + bc[j];
        }
    }
}

extern "C" void kernel_launch(void* const* d_in, const int* in_sizes, int n_in,
                              void* d_out, int out_size, void* d_ws, size_t ws_size,
                              hipStream_t stream) {
    const float* x       = (const float*)d_in[0];
    const float* w_t     = (const float*)d_in[1];
    const float* w_scale = (const float*)d_in[2];
    const float* bias    = (const float*)d_in[3];
    float* out = (float*)d_out;

    const int T = in_sizes[0] / IN_F;

    char* ws = (char*)d_ws;
    signed char* xq = (signed char*)ws;
    signed char* w8 = (signed char*)(ws + (size_t)T * IN_F);
    float* ascale   = (float*)(ws + (size_t)T * IN_F + (size_t)OUT_F * IN_F);

    quantize_act<<<T, 256, 0, stream>>>(x, xq, ascale);

    const int n4 = (OUT_F * IN_F) / 4;
    pack_w<<<(n4 + 255) / 256, 256, 0, stream>>>(w_t, w8, n4);

    const int mblk = T / 32;
    const int nblk = OUT_F / 64;
    const int total_waves = mblk * nblk;
    ternary_gemm<<<total_waves / 8, 256, 0, stream>>>(xq, w8, ascale, w_scale, bias, out,
                                                      nblk);
}
